// SNN_62543313764382
// MI455X (gfx1250) — matrix operand order repaired and checked by reading, not compiled, not run on hardware
//
#include <hip/hip_runtime.h>

typedef __bf16 bf16_t;
typedef __attribute__((ext_vector_type(16))) __bf16 v16bf;
typedef __attribute__((ext_vector_type(8)))  __bf16 v8bf;
typedef __attribute__((ext_vector_type(8)))  float  v8f;
typedef __attribute__((ext_vector_type(4)))  float  f32x4;

#define NB_INPUTS 700
#define NB_HIDDEN 512
#define NB_OUT    20
#define BATCH_N   256
#define NB_STEPS  500
#define KP1       704      // input K padded to multiple of 32
#define KTOT      1216     // 704 (input) + 512 (recurrent)
#define LDS_STRIDE 1224    // bf16 elems per LDS row (row base banks step 36 mod 64)
#define NBLOCKS   64

#define ALPHA_C 0.81873075307798185867f   // exp(-0.01/0.05)
#define BETA_C  0.90483741803595957316f   // exp(-0.01/0.10)

// Build a 16-element A fragment from two contiguous-in-layout 16B LDS chunks;
// union keeps both halves in one 8-VGPR vector (no shuffling copies).
__device__ __forceinline__ v16bf ld_afrag(const bf16_t* p)
{
    union { v16bf v; v8bf h[2]; } af;
    af.h[0] = *(const v8bf*)p;
    af.h[1] = *(const v8bf*)(p + 8);
    return af.v;
}

// ---------------------------------------------------------------------------
// Init: build bf16 transposed weights Wcat^T = [w1 ; 0-pad ; v1]^T ([512][1216],
// N-major) and w2^T padded to 32 cols ([32][512]); zero the barrier counter.
// ---------------------------------------------------------------------------
__global__ __launch_bounds__(256) void snn_init(
    const float* __restrict__ w1, const float* __restrict__ w2,
    const float* __restrict__ v1,
    bf16_t* __restrict__ wcat, bf16_t* __restrict__ w2t,
    unsigned* __restrict__ ctr)
{
    int i = blockIdx.x * 256 + threadIdx.x;
    if (i == 0) *ctr = 0u;
    if (i < NB_HIDDEN * KTOT) {
        int h = i / KTOT, k = i % KTOT;
        float v = 0.0f;
        if (k < NB_INPUTS)      v = w1[k * NB_HIDDEN + h];
        else if (k >= KP1)      v = v1[(k - KP1) * NB_HIDDEN + h];
        wcat[i] = (bf16_t)v;
        return;
    }
    i -= NB_HIDDEN * KTOT;
    if (i < 32 * NB_HIDDEN) {
        int n = i / NB_HIDDEN, k = i % NB_HIDDEN;
        w2t[i] = (bf16_t)((n < NB_OUT) ? w2[k * NB_OUT + n] : 0.0f);
    }
}

// ---------------------------------------------------------------------------
// Device-wide barrier: monotonic counter, one arrival per block per step.
// Release/acquire at agent scope -> global_wb / global_inv so spikes written
// by other WGPs become visible after the barrier.
// ---------------------------------------------------------------------------
__device__ __forceinline__ void grid_barrier(unsigned* ctr, unsigned target)
{
    __threadfence();                 // flush this thread's global stores
    __syncthreads();
    if (threadIdx.x == 0) {
        __hip_atomic_fetch_add(ctr, 1u, __ATOMIC_ACQ_REL, __HIP_MEMORY_SCOPE_AGENT);
        while (__hip_atomic_load(ctr, __ATOMIC_ACQUIRE, __HIP_MEMORY_SCOPE_AGENT) < target) {
            __builtin_amdgcn_s_sleep(2);
        }
    }
    __syncthreads();
    __threadfence();                 // invalidate stale cached lines
}

// ---------------------------------------------------------------------------
// Persistent hidden-layer scan: all 500 time steps in one launch.
// 64 blocks x 256 threads = 512 waves; block = (tileM, 8 consecutive tileN);
// the 8 waves share one 16-row bf16 A tile staged in LDS per step.
// LIF state (syn, mem) lives in registers for the whole 500-step loop.
// GEMM loop: explicit ping-pong double buffering (2x unroll, 19 pairs) so
// fragment loads overlap WMMAs without register-rotation copies.
// ---------------------------------------------------------------------------
__global__ __launch_bounds__(256) void snn_hidden_scan(
    const float* __restrict__ inputs,   // [256][500][700]
    const bf16_t* __restrict__ wcat,    // [512][1216] (N-major)
    float* __restrict__ mem_rec,        // [256][500][512]
    float* __restrict__ spk_rec,        // [256][500][512]
    unsigned* __restrict__ ctr)
{
    __shared__ alignas(16) unsigned lds_u[16 * (LDS_STRIDE / 2)];   // 39168 B
    bf16_t* lds_bf = (bf16_t*)lds_u;

    const int tid   = threadIdx.x;
    const int widx  = tid >> 5;                    // wave in block: 0..7
    const int lane  = tid & 31;
    const int tileM = blockIdx.x >> 2;             // 0..15
    const int tileN = (blockIdx.x & 3) * 8 + widx; // 0..31
    const int hl    = lane >> 4;
    const int ml    = lane & 15;
    const int rowbase = tileM * 16;

    const bf16_t* Brow = wcat + (size_t)(tileN * 16 + ml) * KTOT + (size_t)hl * 16;
    const int colC  = tileN * 16 + ml;
    const bf16_t* Arow = lds_bf + (ml * LDS_STRIDE + hl * 16);

    // LIF state carried in registers across all time steps (exclusive per thread).
    float sreg[8], mreg[8];
#pragma unroll
    for (int i = 0; i < 8; ++i) { sreg[i] = 0.0f; mreg[i] = 0.0f; }

    for (int t = 0; t < NB_STEPS; ++t) {
        if (t > 0) grid_barrier(ctr, (unsigned)t * NBLOCKS);

        // ---- stage A tile: 16 rows x 1216 cols bf16 (packed pair stores) ----
        for (int r = 0; r < 16; ++r) {
            const int row = rowbase + r;
            const float* in_row = inputs + (size_t)row * (NB_STEPS * NB_INPUTS)
                                         + (size_t)t * NB_INPUTS;
            const float* sp_row = (t > 0)
                ? (spk_rec + (size_t)row * (NB_STEPS * NB_HIDDEN)
                           + (size_t)(t - 1) * NB_HIDDEN)
                : (const float*)0;
            for (int p = tid; p < KTOT / 2; p += 256) {
                const int k = 2 * p;                    // pair-aligned regions
                float v0, v1;
                if (k < NB_INPUTS)      { v0 = in_row[k];       v1 = in_row[k + 1]; }
                else if (k < KP1)       { v0 = 0.0f;            v1 = 0.0f;          }
                else if (t > 0)         { v0 = sp_row[k - KP1]; v1 = sp_row[k - KP1 + 1]; }
                else                    { v0 = 0.0f;            v1 = 0.0f;          }
                union { unsigned u; bf16_t h[2]; } pk;
                pk.h[0] = (bf16_t)v0;
                pk.h[1] = (bf16_t)v1;
                lds_u[r * (LDS_STRIDE / 2) + p] = pk.u;
            }
        }
        __syncthreads();

        // ---- GEMM: 38 WMMA k-blocks as 19 ping-pong pairs ----
        v16bf a0 = ld_afrag(Arow);
        v16bf b0 = *(const v16bf*)(Brow);
        v8f c = {};
        for (int k0 = 0; k0 < KTOT; k0 += 64) {
            v16bf a1 = ld_afrag(Arow + k0 + 32);
            v16bf b1 = *(const v16bf*)(Brow + k0 + 32);
            c = __builtin_amdgcn_wmma_f32_16x16x32_bf16(false, a0, false, b0,
                                                        (short)0, c, false, false);
            if (k0 + 64 < KTOT) {               // uniform (scalar) condition
                a0 = ld_afrag(Arow + k0 + 64);
                b0 = *(const v16bf*)(Brow + k0 + 64);
            }
            c = __builtin_amdgcn_wmma_f32_16x16x32_bf16(false, a1, false, b1,
                                                        (short)0, c, false, false);
        }

        // ---- LIF update (registers) + record stores ----
#pragma unroll
        for (int i = 0; i < 8; ++i) {
            const int rowC = rowbase + hl * 8 + i;
            const float so = sreg[i];
            const float mo = mreg[i];
            const float nout = ((mo - 1.0f) > 0.0f) ? 1.0f : 0.0f;
            sreg[i] = ALPHA_C * so + c[i];
            mreg[i] = (BETA_C * mo + so) * (1.0f - nout);
            const size_t ridx = (size_t)rowC * (NB_STEPS * NB_HIDDEN)
                              + (size_t)t * NB_HIDDEN + colC;
            mem_rec[ridx] = mo;     // mem BEFORE update
            spk_rec[ridx] = nout;   // spikes from THIS step
        }
        // (barrier at top of next iteration publishes spk_rec for step t+1)
    }
}

// ---------------------------------------------------------------------------
// Readout GEMM: h2[b*500+t][0..31] = spk_rec[b][t][:] @ w2 (N padded to 32).
// M = 128000, K = 512, N = 32 -> 8000 x 2 tiles, one wave each.
// ---------------------------------------------------------------------------
__global__ __launch_bounds__(256) void snn_readout_gemm(
    const float* __restrict__ spk_rec,  // [B*T][512]
    const bf16_t* __restrict__ w2t,     // [32][512] (N-major)
    float* __restrict__ h2)             // [B*T][32]
{
    const int wave  = (blockIdx.x * 256 + threadIdx.x) >> 5;
    const int lane  = threadIdx.x & 31;
    const int tileM = wave >> 1;
    const int tileN = wave & 1;
    const int hl    = lane >> 4;
    const int ml    = lane & 15;

    const float*  Arow = spk_rec + (size_t)(tileM * 16 + ml) * NB_HIDDEN;
    const bf16_t* Brow = w2t + (size_t)(tileN * 16 + ml) * NB_HIDDEN + (size_t)hl * 16;

    v8f c = {};
    for (int k0 = 0; k0 < NB_HIDDEN; k0 += 32) {
        const int ka = k0 + hl * 16;
        f32x4 a0 = *(const f32x4*)(Arow + ka);
        f32x4 a1 = *(const f32x4*)(Arow + ka + 4);
        f32x4 a2 = *(const f32x4*)(Arow + ka + 8);
        f32x4 a3 = *(const f32x4*)(Arow + ka + 12);
        v16bf a;
#pragma unroll
        for (int j = 0; j < 4; ++j) {
            a[j]      = (bf16_t)a0[j];
            a[4 + j]  = (bf16_t)a1[j];
            a[8 + j]  = (bf16_t)a2[j];
            a[12 + j] = (bf16_t)a3[j];
        }
        v16bf b = *(const v16bf*)(Brow + k0);
        c = __builtin_amdgcn_wmma_f32_16x16x32_bf16(false, a, false, b,
                                                    (short)0, c, false, false);
    }

    const int colC = tileN * 16 + ml;
#pragma unroll
    for (int i = 0; i < 8; ++i) {
        const int rowC = tileM * 16 + hl * 8 + i;
        h2[(size_t)rowC * 32 + colC] = c[i];
    }
}

// ---------------------------------------------------------------------------
// Readout leaky-integrator scan: one thread per (batch, output) pair.
// ---------------------------------------------------------------------------
__global__ __launch_bounds__(256) void snn_readout_scan(
    const float* __restrict__ h2,       // [B*T][32]
    float* __restrict__ out_rec)        // [B][501][20]
{
    const int idx = blockIdx.x * 256 + threadIdx.x;   // 0..8191
    const int b = idx >> 5;
    const int o = idx & 31;
    if (o >= NB_OUT) return;

    float* orow = out_rec + (size_t)b * ((NB_STEPS + 1) * NB_OUT);
    orow[o] = 0.0f;                                   // t = 0 row
    const float* hrow = h2 + (size_t)b * NB_STEPS * 32 + o;

    float flt = 0.0f, outv = 0.0f;
    for (int t = 0; t < NB_STEPS; ++t) {
        const float h  = hrow[(size_t)t * 32];
        const float nf = ALPHA_C * flt + h;
        const float no = BETA_C * outv + flt;
        orow[(size_t)(t + 1) * NB_OUT + o] = no;
        flt = nf;
        outv = no;
    }
}

// ---------------------------------------------------------------------------
// Workspace layout (bytes):
//   wcat  [512*1216 bf16]    @ 0         (1245184)
//   w2t   [32*512 bf16]      @ 1245184   (32768)
//   ctr   [unsigned]         @ 1277952   (256)
//   h2    [128000*32 f32]    @ 1278208   (16384000)  -> total 17662208 B
// ---------------------------------------------------------------------------
extern "C" void kernel_launch(void* const* d_in, const int* in_sizes, int n_in,
                              void* d_out, int out_size, void* d_ws, size_t ws_size,
                              hipStream_t stream) {
    (void)in_sizes; (void)n_in; (void)out_size; (void)ws_size;

    const float* inputs = (const float*)d_in[0];   // [256][500][700]
    const float* w1     = (const float*)d_in[1];   // [700][512]
    const float* w2     = (const float*)d_in[2];   // [512][20]
    const float* v1     = (const float*)d_in[3];   // [512][512]

    float* out_rec = (float*)d_out;                                        // [256][501][20]
    float* mem_rec = out_rec + (size_t)BATCH_N * (NB_STEPS + 1) * NB_OUT;  // [256][500][512]
    float* spk_rec = mem_rec + (size_t)BATCH_N * NB_STEPS * NB_HIDDEN;     // [256][500][512]

    char*     ws   = (char*)d_ws;
    bf16_t*   wcat = (bf16_t*)(ws);
    bf16_t*   w2t  = (bf16_t*)(ws + 1245184);
    unsigned* ctr  = (unsigned*)(ws + 1277952);
    float*    h2   = (float*)(ws + 1278208);

    // 622592 (wcat) + 16384 (w2t) = 638976 threads
    snn_init<<<2496, 256, 0, stream>>>(w1, w2, v1, wcat, w2t, ctr);

    // One persistent launch replaces 500 per-step launches.
    snn_hidden_scan<<<NBLOCKS, 256, 0, stream>>>(inputs, wcat,
                                                 mem_rec, spk_rec, ctr);

    snn_readout_gemm<<<2000, 256, 0, stream>>>(spk_rec, w2t, h2);
    snn_readout_scan<<<32, 256, 0, stream>>>(h2, out_rec);
}
